// Model_4964982194729
// MI455X (gfx1250) — matrix operand order repaired and checked by reading, not compiled, not run on hardware
//
#include <hip/hip_runtime.h>
#include <hip/hip_bf16.h>
#include <math.h>

#define BB 512
#define TT 1024
#define HH 64
#define NC 1098

typedef __attribute__((ext_vector_type(16))) __bf16 v16bf;
typedef __attribute__((ext_vector_type(8)))  __bf16 v8bf;
typedef __attribute__((ext_vector_type(8)))  float  v8f;

__device__ __forceinline__ float sigmoidf_(float x) { return 1.0f / (1.0f + __expf(-x)); }
__device__ __forceinline__ float tanhf_(float x)    { return 1.0f - 2.0f / (__expf(2.0f * x) + 1.0f); }

// generic (flat) shared pointer -> 32-bit LDS offset for async-to-LDS vdst operand
__device__ __forceinline__ unsigned lds_offset(const void* p) {
  return (unsigned)(size_t)(__attribute__((address_space(3))) const char*)(const char*)p;
}

// A-operand (16-bit, 16x32): lane halves hold K chunks {k0+8h..+8, k0+16+8h..+8} (ISA 7.12.2)
__device__ __forceinline__ v16bf load_a16(const __bf16* row, int k0, int half) {
  v8bf a = *(const v8bf*)(row + k0 + 8 * half);
  v8bf b = *(const v8bf*)(row + k0 + 16 + 8 * half);
  return __builtin_shufflevector(a, b, 0, 1, 2, 3, 4, 5, 6, 7, 8, 9, 10, 11, 12, 13, 14, 15);
}

// B-operand (16-bit, 32x16): lane = N column, lane halves hold K chunks {k0+8h..+8, k0+16+8h..+8}
__device__ __forceinline__ v16bf load_b16(const __bf16* colrow, int k0, int half) {
  v8bf a = *(const v8bf*)(colrow + k0 + 8 * half);
  v8bf b = *(const v8bf*)(colrow + k0 + 16 + 8 * half);
  return __builtin_shufflevector(a, b, 0, 1, 2, 3, 4, 5, 6, 7, 8, 9, 10, 11, 12, 13, 14, 15);
}

__device__ __forceinline__ v8f wmma_bf16(v16bf a, v16bf b, v8f c) {
  return __builtin_amdgcn_wmma_f32_16x16x32_bf16(false, a, false, b, (short)0, c, false, false);
}

// One bidirectional LSTM layer. grid = (B/16, 2 dirs), block = 128 (4 waves).
// Wave sw owns hidden units [16sw,16sw+16): its 4 gate accumulators (i,f,g,o)
// cover gate columns 64g+16sw+lm, so the c/h elementwise update is lane-local.
template <int LAYER>
__global__ void lstm_layer_kernel(
    const float* __restrict__ x4,          // LAYER==0: [B,T,4]
    const __bf16* __restrict__ xin,        // LAYER==1: [B,T,128]
    const int* __restrict__ lengths,       // [B]
    const float* __restrict__ Wih_f, const float* __restrict__ Whh_f,
    const float* __restrict__ bih_f, const float* __restrict__ bhh_f,
    const float* __restrict__ Wih_b, const float* __restrict__ Whh_b,
    const float* __restrict__ bih_b, const float* __restrict__ bhh_b,
    __bf16* __restrict__ hseq,             // LAYER==0 out: [B,T,128] (fwd|bwd-reversed)
    float* __restrict__ hlast)             // LAYER==1 out: [B,128]
{
  const int b0   = blockIdx.x * 16;
  const int dir  = blockIdx.y;             // 0 = forward, 1 = backward
  const int tid  = threadIdx.x;
  const int sw   = tid >> 5;               // wave id = unit tile 0..3
  const int lane = tid & 31;
  const int half = lane >> 4;
  const int lm   = lane & 15;

  const float* Wih = dir ? Wih_b : Wih_f;
  const float* Whh = dir ? Whh_b : Whh_f;
  const float* bih = dir ? bih_b : bih_f;
  const float* bhh = dir ? bhh_b : bhh_f;

  extern __shared__ char smem_raw[];
  // [ sX double-buffer (layer1 only) | sWih (layer1 only) | sWhh | sBias | sH | sLen ]
  __bf16* sX    = (__bf16*)smem_raw;                       // [2][16][128] bf16
  __bf16* sWih  = sX + (LAYER == 1 ? 2 * 16 * 128 : 0);    // [256][128] bf16
  __bf16* sWhh  = sWih + (LAYER == 1 ? 256 * 128 : 0);     // [256][64]  bf16
  float*  sBias = (float*)(sWhh + 256 * 64);               // [256] f32
  __bf16* sH    = (__bf16*)(sBias + 256);                  // [16][64] bf16
  int*    sLen  = (int*)(sH + 16 * 64);                    // [16]

  if (LAYER == 1)
    for (int i = tid; i < 256 * 128; i += blockDim.x) sWih[i] = (__bf16)Wih[i];
  for (int i = tid; i < 256 * 64; i += blockDim.x) sWhh[i] = (__bf16)Whh[i];
  for (int i = tid; i < 256; i += blockDim.x) sBias[i] = bih[i] + bhh[i];
  for (int i = tid; i < 16 * 64; i += blockDim.x) sH[i] = (__bf16)0.0f;
  if (tid < 16) sLen[tid] = lengths[b0 + tid];
  __syncthreads();

  const int u = 16 * sw + lm;              // this lane's hidden unit (N of C/D tile)

  // loop-invariant per-lane state: biases, lengths
  float bI = sBias[u], bF = sBias[64 + u], bG = sBias[128 + u], bO = sBias[192 + u];
  int lenm[8];
#pragma unroll
  for (int r = 0; r < 8; ++r) lenm[r] = sLen[r + 8 * half];

  // async x-tile copy assignment (layer1): thread copies row=tid>>3, 2x16B chunks
  const int xrow_id = tid >> 3;            // 0..15
  const int xsub    = tid & 7;             // 0..7 -> bytes xsub*16 and xsub*16+128
  const int lenRow  = (LAYER == 1) ? sLen[xrow_id] : 0;

  // layer0: keep the K=4 input weights in registers (row = this lane's gate col)
  float wih0[4][4];
  if (LAYER == 0) {
#pragma unroll
    for (int g = 0; g < 4; ++g) {
      const float* wr = Wih + (size_t)(64 * g + u) * 4;
#pragma unroll
      for (int k = 0; k < 4; ++k) wih0[g][k] = wr[k];
    }
  }

  float cst[8];
#pragma unroll
  for (int r = 0; r < 8; ++r) cst[r] = 0.0f;

  // prologue: async-stage x tile for t=0 into buffer 0 (layer1)
  if (LAYER == 1) {
    int tx0 = dir ? lenRow - 1 : 0;        // rev(0) = len-1 (len >= 1)
    unsigned long long g =
        (unsigned long long)(size_t)(xin + ((size_t)(b0 + xrow_id) * TT + tx0) * 128) +
        (unsigned)(xsub * 16);
    unsigned l = lds_offset(sX) + (unsigned)(xrow_id * 256 + xsub * 16);
    asm volatile("global_load_async_to_lds_b128 %0, %1, off" :: "v"(l), "v"(g) : "memory");
    asm volatile("global_load_async_to_lds_b128 %0, %1, off offset:128" :: "v"(l), "v"(g) : "memory");
  }

  for (int t = 0; t < TT; ++t) {
    if (LAYER == 1) {
      if (t + 1 < TT) {                    // issue tile t+1 into the other buffer
        int t1 = t + 1;
        int tx1 = dir ? (t1 < lenRow ? lenRow - 1 - t1 : t1) : t1;
        unsigned long long g =
            (unsigned long long)(size_t)(xin + ((size_t)(b0 + xrow_id) * TT + tx1) * 128) +
            (unsigned)(xsub * 16);
        unsigned l = lds_offset(sX) + (unsigned)((t1 & 1) * 4096 + xrow_id * 256 + xsub * 16);
        asm volatile("global_load_async_to_lds_b128 %0, %1, off" :: "v"(l), "v"(g) : "memory");
        asm volatile("global_load_async_to_lds_b128 %0, %1, off offset:128" :: "v"(l), "v"(g) : "memory");
        // retire tile t (in-order): only the 2 just-issued ops may remain outstanding
        asm volatile("s_wait_asynccnt 0x2" ::: "memory");
      } else {
        asm volatile("s_wait_asynccnt 0x0" ::: "memory");
      }
    }

    __syncthreads();                       // sH from prev step + async x tile visible

    v16bf aH0 = load_a16(&sH[lm * 64], 0, half);
    v16bf aH1 = load_a16(&sH[lm * 64], 32, half);

    v16bf aX[4];
    if (LAYER == 1) {
      const __bf16* xrow = sX + (size_t)((t & 1) * 2048 + lm * 128);
#pragma unroll
      for (int c = 0; c < 4; ++c) aX[c] = load_a16(xrow, 32 * c, half);
    }

    v8f acc[4] = {};                       // bias folded into elementwise stage

    if (LAYER == 0) {                      // scalar K=4 input contribution
#pragma unroll
      for (int r = 0; r < 8; ++r) {
        int m = r + 8 * half;
        int len = lenm[r];
        int tx = dir ? (t < len ? len - 1 - t : t) : t;
        const float* xp = x4 + ((size_t)(b0 + m) * TT + tx) * 4;
        float x0 = xp[0], x1 = xp[1], x2 = xp[2], x3 = xp[3];
#pragma unroll
        for (int g = 0; g < 4; ++g)
          acc[g][r] = wih0[g][0] * x0 + wih0[g][1] * x1 + wih0[g][2] * x2 + wih0[g][3] * x3;
      }
    }

#pragma unroll
    for (int g = 0; g < 4; ++g) {
      int ncol = 64 * g + u;
      if (LAYER == 1) {
        const __bf16* wrow = sWih + (size_t)ncol * 128;
#pragma unroll
        for (int c = 0; c < 4; ++c)
          acc[g] = wmma_bf16(aX[c], load_b16(wrow, 32 * c, half), acc[g]);
      }
      const __bf16* hrow = sWhh + (size_t)ncol * 64;
      acc[g] = wmma_bf16(aH0, load_b16(hrow, 0, half), acc[g]);
      acc[g] = wmma_bf16(aH1, load_b16(hrow, 32, half), acc[g]);
    }

    float hval[8];
#pragma unroll
    for (int r = 0; r < 8; ++r) {
      float ig = sigmoidf_(acc[0][r] + bI);
      float fg = sigmoidf_(acc[1][r] + bF);
      float gg = tanhf_(acc[2][r] + bG);
      float og = sigmoidf_(acc[3][r] + bO);
      float cn = fg * cst[r] + ig * gg;
      cst[r] = cn;
      hval[r] = og * tanhf_(cn);
    }

    __syncthreads();                       // all reads of sH done before overwrite

#pragma unroll
    for (int r = 0; r < 8; ++r) {
      int m = r + 8 * half;                // C/D layout: M = r + 8*half, N = lane&15
      sH[m * 64 + u] = (__bf16)hval[r];
      if (LAYER == 0) {
        int len = lenm[r];
        int ts = dir ? (t < len ? len - 1 - t : t) : t;   // bwd stores reversed back
        hseq[((size_t)(b0 + m) * TT + ts) * 128 + dir * 64 + u] = (__bf16)hval[r];
      } else {
        if (!dir && t == lenm[r] - 1) hlast[(size_t)(b0 + m) * 128 + u] = hval[r];
        if (dir && t == 0)            hlast[(size_t)(b0 + m) * 128 + 64 + u] = hval[r];
      }
    }
  }
}

// Final classifier: [512,128] x [128,1098] + bias, one 16x16 tile per wave.
// Out-of-range columns read a clamped row (garbage result) and are masked at the store.
__global__ void classifier_kernel(const float* __restrict__ hlast,
                                  const float* __restrict__ Wout,   // [NC,128]
                                  const float* __restrict__ bout,   // [NC]
                                  float* __restrict__ out)          // [B,NC]
{
  const int m0 = blockIdx.x * 16, n0 = blockIdx.y * 16;
  const int lane = threadIdx.x & 31;
  const int half = lane >> 4, lm = lane & 15;
  const int n = n0 + lm;
  const bool nval = n < NC;
  const int nc = nval ? n : NC - 1;        // clamped, always-valid index

  v16bf aA[4];
  const float* arow = hlast + (size_t)(m0 + lm) * 128;
#pragma unroll
  for (int c = 0; c < 4; ++c) {
    v16bf v;
#pragma unroll
    for (int i = 0; i < 8; ++i) {
      v[i]     = (__bf16)arow[32 * c + 8 * half + i];
      v[8 + i] = (__bf16)arow[32 * c + 16 + 8 * half + i];
    }
    aA[c] = v;
  }

  v8f acc;
  float bi = bout[nc];
#pragma unroll
  for (int r = 0; r < 8; ++r) acc[r] = bi;

  const float* wrow = Wout + (size_t)nc * 128;
#pragma unroll
  for (int c = 0; c < 4; ++c) {
    v16bf bv;
#pragma unroll
    for (int i = 0; i < 16; ++i) bv[i] = (__bf16)wrow[32 * c + (i < 8 ? 8 * half + i : 16 + 8 * half + (i - 8))];
    acc = wmma_bf16(aA[c], bv, acc);
  }

#pragma unroll
  for (int r = 0; r < 8; ++r) {
    int m = r + 8 * half;
    if (nval) out[(size_t)(m0 + m) * NC + n] = acc[r];
  }
}

extern "C" void kernel_launch(void* const* d_in, const int* in_sizes, int n_in,
                              void* d_out, int out_size, void* d_ws, size_t ws_size,
                              hipStream_t stream) {
  const float* x        = (const float*)d_in[0];
  const int*   lengths  = (const int*)d_in[1];
  const float* Wih_l0f = (const float*)d_in[2],  *Whh_l0f = (const float*)d_in[3];
  const float* bih_l0f = (const float*)d_in[4],  *bhh_l0f = (const float*)d_in[5];
  const float* Wih_l0b = (const float*)d_in[6],  *Whh_l0b = (const float*)d_in[7];
  const float* bih_l0b = (const float*)d_in[8],  *bhh_l0b = (const float*)d_in[9];
  const float* Wih_l1f = (const float*)d_in[10], *Whh_l1f = (const float*)d_in[11];
  const float* bih_l1f = (const float*)d_in[12], *bhh_l1f = (const float*)d_in[13];
  const float* Wih_l1b = (const float*)d_in[14], *Whh_l1b = (const float*)d_in[15];
  const float* bih_l1b = (const float*)d_in[16], *bhh_l1b = (const float*)d_in[17];
  const float* Wout    = (const float*)d_in[18], *bout    = (const float*)d_in[19];

  // workspace: h0 activations [B,T,128] bf16 (128 MB) + hlast [B,128] f32
  __bf16* h0    = (__bf16*)d_ws;
  float*  hlast = (float*)((char*)d_ws + (size_t)BB * TT * 128 * sizeof(__bf16));

  const size_t lds0 = (size_t)256 * 64 * 2 + 256 * 4 + 16 * 64 * 2 + 16 * 4;
  const size_t lds1 = lds0 + (size_t)2 * 16 * 128 * 2 + (size_t)256 * 128 * 2;

  dim3 grid(BB / 16, 2), block(128);
  lstm_layer_kernel<0><<<grid, block, lds0, stream>>>(
      x, nullptr, lengths,
      Wih_l0f, Whh_l0f, bih_l0f, bhh_l0f,
      Wih_l0b, Whh_l0b, bih_l0b, bhh_l0b,
      h0, nullptr);
  lstm_layer_kernel<1><<<grid, block, lds1, stream>>>(
      nullptr, h0, lengths,
      Wih_l1f, Whh_l1f, bih_l1f, bhh_l1f,
      Wih_l1b, Whh_l1b, bih_l1b, bhh_l1b,
      nullptr, hlast);
  classifier_kernel<<<dim3(BB / 16, (NC + 15) / 16), dim3(32), 0, stream>>>(
      hlast, Wout, bout, (float*)d_out);
}
